// DCKModule_69148973466345
// MI455X (gfx1250) — matrix operand order repaired and checked by reading, not compiled, not run on hardware
//
#include <hip/hip_runtime.h>

typedef __attribute__((ext_vector_type(16))) _Float16 v16h;
typedef __attribute__((ext_vector_type(8)))  _Float16 v8h;
typedef __attribute__((ext_vector_type(8)))  float    v8f;

#define KSZ   7
#define PADR  3
#define GRP   16
#define CIN   256
#define RCH   64
#define IMGW  96
#define HW    9216           // 96*96
#define BATCH 8
#define NPIX  (BATCH * HW)   // 73728
#define NTAP  49
#define NDF   (GRP * NTAP)   // 784

union F16x16 { v16h v; v8h h2[2]; _Float16 h[16]; };

// K index inside a 16-bit A-fragment (16x32, ISA 7.12.2):
// lanes 0-15: VGPR0..3 -> K{0..7}, VGPR4..7 -> K{16..23}; lanes 16-31: +8
__device__ __forceinline__ int ka_idx(int h, int sel) {
    return ((h & 8) << 1) | (sel << 3) | (h & 7);
}

// ---------------------------------------------------------------------------
// Kernel 0: one-time prep — f32->f16 of W1 (64x256) and W2 (784x64), plus
// folded BN scale/shift (64 floats each) so the GEMM epilogue is FMA-only.
// ---------------------------------------------------------------------------
__global__ __launch_bounds__(256) void prep_weights(
    const float* __restrict__ W1, const float* __restrict__ W2,
    const float* __restrict__ gamma, const float* __restrict__ beta,
    const float* __restrict__ mean,  const float* __restrict__ var,
    _Float16* __restrict__ w1h, _Float16* __restrict__ w2h,
    float* __restrict__ bnsc, float* __restrict__ bnsh)
{
    int i = blockIdx.x * blockDim.x + threadIdx.x;
    if (i < RCH * CIN) w1h[i] = (_Float16)W1[i];
    if (i < NDF * RCH) w2h[i] = (_Float16)W2[i];
    if (i < RCH) {
        float sc = gamma[i] * rsqrtf(var[i] + 1e-5f);
        bnsc[i] = sc;
        bnsh[i] = beta[i] - mean[i] * sc;
    }
}

// ---------------------------------------------------------------------------
// Kernel 1: x = ReLU(BN(W1 @ guide)), stored f16 as [pixel][64]
// 128 threads = 4 waves; each wave: 16-pixel M-tile x N=64, K=256
// ---------------------------------------------------------------------------
__global__ __launch_bounds__(128) void stage1_gemm_bn_relu(
    const float* __restrict__ guide,
    const _Float16* __restrict__ w1h,
    const float* __restrict__ bnsc, const float* __restrict__ bnsh,
    _Float16* __restrict__ xout)
{
    const int wave = threadIdx.x >> 5;
    const int lane = threadIdx.x & 31;
    const int row  = lane & 15;
    const int sel  = lane >> 4;
    const int pt   = blockIdx.x * 64 + wave * 16;   // wave's pixel tile base
    const int pa   = pt + row;                      // this lane's A-row pixel
    const int bidx = pa / HW;
    const int phw  = pa - bidx * HW;
    const float* gbase = guide + (size_t)bidx * CIN * HW + phw;

    v8f acc[4] = {};
    for (int kb = 0; kb < 8; ++kb) {
        const int k0 = kb * 32;
        F16x16 a;
        #pragma unroll
        for (int h = 0; h < 16; ++h)
            a.h[h] = (_Float16)gbase[(size_t)(k0 + ka_idx(h, sel)) * HW];
        #pragma unroll
        for (int n = 0; n < 4; ++n) {
            const int o = n * 16 + row;             // B column = lane&15
            F16x16 b;
            b.h2[0] = *(const v8h*)(w1h + o * CIN + k0 + sel * 8);
            b.h2[1] = *(const v8h*)(w1h + o * CIN + k0 + 16 + sel * 8);
            acc[n] = __builtin_amdgcn_wmma_f32_16x16x32_f16(
                         false, a.v, false, b.v, (short)0, acc[n], false, false);
        }
    }
    // BN + ReLU epilogue; C layout: lane->N, VGPR j -> M = j+8*sel
    #pragma unroll
    for (int n = 0; n < 4; ++n) {
        const int o = n * 16 + row;
        const float sc = bnsc[o];
        const float sh = bnsh[o];
        #pragma unroll
        for (int j = 0; j < 8; ++j) {
            const int p = pt + j + 8 * sel;
            float v = acc[n][j] * sc + sh;
            v = v > 0.f ? v : 0.f;
            xout[(size_t)p * RCH + o] = (_Float16)v;
        }
    }
}

// ---------------------------------------------------------------------------
// Kernel 2: fused  df = W2 @ x  (WMMA -> LDS)  +  involution + residual
// 64 threads = 2 waves, 32-pixel tile per workgroup. df tile: 32x784 f16.
// ---------------------------------------------------------------------------
__global__ __launch_bounds__(64) void stage2_df_involution(
    const _Float16* __restrict__ xin,
    const _Float16* __restrict__ w2h,
    const float* __restrict__ feat,
    float* __restrict__ out)
{
    __shared__ _Float16 df[32 * NDF];   // 50,176 bytes
    const int wave = threadIdx.x >> 5;
    const int lane = threadIdx.x & 31;
    const int row  = lane & 15;
    const int sel  = lane >> 4;
    const int pt   = blockIdx.x * 32;
    const int pw   = pt + wave * 16;

    // A fragments: this wave's 16 pixels, K = 64 as two 32-wide slices.
    // ka_idx ranges split into two contiguous 8-half runs -> 16B vector loads.
    const _Float16* xp = xin + (size_t)(pw + row) * RCH;
    F16x16 a0, a1;
    a0.h2[0] = *(const v8h*)(xp + sel * 8);
    a0.h2[1] = *(const v8h*)(xp + 16 + sel * 8);
    a1.h2[0] = *(const v8h*)(xp + 32 + sel * 8);
    a1.h2[1] = *(const v8h*)(xp + 48 + sel * 8);

    // 49 N-chunks of 16 cover all 784 dynamic-filter outputs
    for (int chunk = 0; chunk < NTAP; ++chunk) {
        const int o = chunk * 16 + row;
        const _Float16* wp = w2h + o * RCH;
        F16x16 b0, b1;
        b0.h2[0] = *(const v8h*)(wp + sel * 8);
        b0.h2[1] = *(const v8h*)(wp + 16 + sel * 8);
        b1.h2[0] = *(const v8h*)(wp + 32 + sel * 8);
        b1.h2[1] = *(const v8h*)(wp + 48 + sel * 8);
        v8f acc = {};
        acc = __builtin_amdgcn_wmma_f32_16x16x32_f16(
                  false, a0.v, false, b0.v, (short)0, acc, false, false);
        acc = __builtin_amdgcn_wmma_f32_16x16x32_f16(
                  false, a1.v, false, b1.v, (short)0, acc, false, false);
        #pragma unroll
        for (int j = 0; j < 8; ++j) {
            const int pl = wave * 16 + j + 8 * sel;
            df[pl * NDF + o] = (_Float16)acc[j];
        }
    }
    __syncthreads();

    // Involution + residual: 32 px * 256 ch / 64 threads = 128 iterations.
    // pl varies fastest across lanes -> coalesced feature/out accesses.
    for (int it = 0; it < 128; ++it) {
        const int idx  = it * 64 + threadIdx.x;
        const int pl   = idx & 31;
        const int c    = idx >> 5;
        const int p    = pt + pl;
        const int bidx = p / HW;
        const int phw  = p - bidx * HW;
        const int y    = phw / IMGW;
        const int x0   = phw - y * IMGW;
        const int g    = c >> 4;
        const _Float16* dfp = df + pl * NDF + g * NTAP;
        const float* fb = feat + ((size_t)bidx * CIN + c) * HW;
        float s = 0.f;
        if (y >= PADR && y < IMGW - PADR && x0 >= PADR && x0 < IMGW - PADR) {
            // interior fast path: all 49 taps in-bounds; immediate-offset loads
            const float* fc = fb + (y - PADR) * IMGW + (x0 - PADR);
            #pragma unroll
            for (int di = 0; di < KSZ; ++di) {
                #pragma unroll
                for (int dj = 0; dj < KSZ; ++dj) {
                    s += (float)dfp[di * KSZ + dj] * fc[di * IMGW + dj];
                }
            }
        } else {
            // border path: per-tap bounds checks (zero padding)
            #pragma unroll
            for (int di = 0; di < KSZ; ++di) {
                const int yy = y + di - PADR;
                if (yy < 0 || yy >= IMGW) continue;
                #pragma unroll
                for (int dj = 0; dj < KSZ; ++dj) {
                    const int xx = x0 + dj - PADR;
                    if (xx < 0 || xx >= IMGW) continue;
                    s += (float)dfp[di * KSZ + dj] * fb[yy * IMGW + xx];
                }
            }
        }
        out[((size_t)bidx * CIN + c) * HW + phw] = s + fb[phw];
    }
}

// ---------------------------------------------------------------------------
extern "C" void kernel_launch(void* const* d_in, const int* in_sizes, int n_in,
                              void* d_out, int out_size, void* d_ws, size_t ws_size,
                              hipStream_t stream) {
    (void)in_sizes; (void)n_in; (void)out_size; (void)ws_size;
    const float* feat  = (const float*)d_in[0];
    const float* guide = (const float*)d_in[1];
    const float* W1    = (const float*)d_in[2];
    const float* gamma = (const float*)d_in[3];
    const float* beta  = (const float*)d_in[4];
    const float* mean  = (const float*)d_in[5];
    const float* var   = (const float*)d_in[6];
    const float* W2    = (const float*)d_in[7];
    float* out = (float*)d_out;

    // ws layout: x f16[73728*64] | W1h f16[16384] | W2h f16[50176]
    //            | bnsc f32[64] | bnsh f32[64]          (~9.6 MB total)
    _Float16* xbuf = (_Float16*)d_ws;
    _Float16* w1h  = xbuf + (size_t)NPIX * RCH;
    _Float16* w2h  = w1h + RCH * CIN;
    float*    bnsc = (float*)(w2h + NDF * RCH);
    float*    bnsh = bnsc + RCH;

    prep_weights<<<(NDF * RCH + 255) / 256, 256, 0, stream>>>(
        W1, W2, gamma, beta, mean, var, w1h, w2h, bnsc, bnsh);
    stage1_gemm_bn_relu<<<NPIX / 64, 128, 0, stream>>>(guide, w1h, bnsc, bnsh, xbuf);
    stage2_df_involution<<<NPIX / 32, 64, 0, stream>>>(xbuf, w2h, feat, out);
}
